// Target_Attention_89936615178797
// MI455X (gfx1250) — matrix operand order repaired and checked by reading, not compiled, not run on hardware
//
#include <hip/hip_runtime.h>

// Target attention fused kernel for gfx1250 (MI455X).
// B=256, S=512, N=256, D1=D2=256.
// Per workgroup: one (batch, 64-wide n tile). 256 threads = 8 wave32.
// All GEMMs via v_wmma_f32_16x16x32_bf16 (f32 accumulate), softmax in f32.

typedef __attribute__((ext_vector_type(16))) __bf16 v16bf;
typedef __attribute__((ext_vector_type(8)))  __bf16 v8bf;
typedef __attribute__((ext_vector_type(8)))  float  v8f;
typedef __attribute__((ext_vector_type(4)))  float  v4f;

#define WMMA_BF16(a, b, c) \
  __builtin_amdgcn_wmma_f32_16x16x32_bf16(false, (a), false, (b), (short)0, (c), false, false)

__device__ __forceinline__ v16bf cat16(v8bf lo, v8bf hi) {
  return __builtin_shufflevector(lo, hi, 0,1,2,3,4,5,6,7,8,9,10,11,12,13,14,15);
}

// A-fragment (16x32 bf16, M x K) from LDS stored row-major [M][K], stride in halfs.
// Layout (ISA 7.12.2): lane holds M = lane&15; halves 0..7 -> K = kb+{0..7},
// halves 8..15 -> K = 16+kb+{0..7}, kb = 8*(lane>=16).
__device__ __forceinline__ v16bf ldsA(const __bf16* base, int lane, int strideH,
                                      int m0, int k0) {
  int m  = m0 + (lane & 15);
  int kb = (lane >> 4) << 3;                 // 0 or 8
  const __bf16* p = base + (size_t)m * strideH + k0 + kb;
  v8bf lo = *(const v8bf*)p;                 // K = k0+kb .. +7   (16B aligned)
  v8bf hi = *(const v8bf*)(p + 16);          // K = k0+16+kb .. +7
  return cat16(lo, hi);
}

// B-fragment (32x16 bf16, K x N) from LDS stored row-major [N][K] (i.e. B^T),
// stride in halfs. Layout: lane holds N = lane&15; halves 0..7 -> K = k0+kb+{0..7},
// halves 8..15 -> K = k0+16+kb+{0..7}, kb = 8*(lane>=16).
__device__ __forceinline__ v16bf ldsB(const __bf16* base, int lane, int strideH,
                                      int n0, int k0) {
  int n  = n0 + (lane & 15);
  int kb = (lane >> 4) << 3;                 // 0 or 8
  const __bf16* p = base + (size_t)n * strideH + k0 + kb;
  v8bf lo = *(const v8bf*)p;
  v8bf hi = *(const v8bf*)(p + 16);
  return cat16(lo, hi);
}

// ---- dynamic LDS layout (bytes) ----
// stage   @ 0       : max(64x264 bf16 = 33792, 256x72 bf16 = 36864) -> 36864
// tgtW    @ 36864   : 64x264 bf16  = 33792   (n-major, padded)
// score   @ 70656   : 512x68 f32   = 139264  (s-major, padded)  [W-chunk overlays here in phase 1]
// weightT @ 209920  : 64x528 bf16  = 67584   (n-major, padded)
// red     @ 277504  : 4x64 f32; cmax @ 278528: 64 f32; csum @ 278784: 64 f32
#define SMEM_BYTES 279040

__global__ __launch_bounds__(256)
void ta_fused_kernel(const float* __restrict__ seq, const float* __restrict__ tgt,
                     const unsigned char* __restrict__ mask,
                     const float* __restrict__ W, float* __restrict__ out) {
  extern __shared__ char smem[];
  __bf16* s_stage = (__bf16*)(smem);
  __bf16* s_tgtw  = (__bf16*)(smem + 36864);
  float*  s_score = (float*) (smem + 70656);
  __bf16* s_wch   = (__bf16*)(smem + 70656);    // overlay: W chunk [256][40] bf16
  __bf16* s_wT    = (__bf16*)(smem + 209920);
  float*  s_red   = (float*) (smem + 277504);
  float*  s_cmax  = (float*) (smem + 278528);
  float*  s_csum  = (float*) (smem + 278784);

  const int tid  = threadIdx.x;
  const int lane = tid & 31;
  const int w    = tid >> 5;
  const int b    = blockIdx.y;
  const int nb   = blockIdx.x * 64;

  const float* seqb = seq + (size_t)b * 512 * 256;
  const float* tgtb = tgt + ((size_t)b * 256 + nb) * 256;
  const unsigned char* maskb = mask + (size_t)b * 512;   // jax bool -> 1 byte/elem

  // ================= Phase 1: tgtW[n][d] = sum_e tgt[n][e] * W[d][e] =========
  // stage target tile 64x256 f32 -> bf16 LDS [64][264]
  #pragma unroll 4
  for (int j = 0; j < 16; ++j) {
    int i4 = tid + j * 256;                  // 4096 float4
    int r = i4 >> 6, c4 = i4 & 63;
    v4f v = *(const v4f*)(tgtb + r * 256 + c4 * 4);
    __bf16* dst = s_stage + r * 264 + c4 * 4;
    dst[0] = (__bf16)v.x; dst[1] = (__bf16)v.y;
    dst[2] = (__bf16)v.z; dst[3] = (__bf16)v.w;
  }

  const int nstrip1 = w & 3;                 // n-rows nstrip1*16..+15
  const int dblk    = (w >> 2) * 8;          // d-tiles dblk..dblk+7
  v8f acc1[8];
  #pragma unroll
  for (int t = 0; t < 8; ++t) acc1[t] = (v8f){0.f,0.f,0.f,0.f,0.f,0.f,0.f,0.f};

  for (int es = 0; es < 8; ++es) {           // K over e, 8 x 32
    // stage W chunk [256 d][32 e] f32 -> bf16 LDS [256][40]
    #pragma unroll 4
    for (int j = 0; j < 8; ++j) {
      int i4 = tid + j * 256;                // 2048 float4
      int d = i4 >> 3, e4 = i4 & 7;
      v4f v = *(const v4f*)(W + d * 256 + es * 32 + e4 * 4);
      __bf16* dst = s_wch + d * 40 + e4 * 4;
      dst[0] = (__bf16)v.x; dst[1] = (__bf16)v.y;
      dst[2] = (__bf16)v.z; dst[3] = (__bf16)v.w;
    }
    __syncthreads();
    v16bf a = ldsA(s_stage, lane, 264, nstrip1 * 16, es * 32);
    #pragma unroll
    for (int t = 0; t < 8; ++t) {
      v16bf bf = ldsB(s_wch, lane, 40, (dblk + t) * 16, 0);
      acc1[t] = WMMA_BF16(a, bf, acc1[t]);
    }
    __syncthreads();
  }
  {
    // D layout: lane -> col N = lane&15, VGPR r -> row M = r + 8*(lane>=16)
    int col = lane & 15, rofs = (lane >> 4) * 8;
    #pragma unroll
    for (int t = 0; t < 8; ++t) {
      int d = (dblk + t) * 16 + col;
      #pragma unroll
      for (int r = 0; r < 8; ++r)
        s_tgtw[(nstrip1 * 16 + rofs + r) * 264 + d] = (__bf16)acc1[t][r];
    }
  }
  __syncthreads();

  // ================= Phase 2: score[s][n] = seq[s][:] . tgtW[n][:] ===========
  const int sstrip = w >> 1;                 // 0..3
  const int npair  = (w & 1) * 2;            // n-tiles npair, npair+1
  for (int sc = 0; sc < 8; ++sc) {
    int sbase = sc * 64;
    #pragma unroll 4
    for (int j = 0; j < 16; ++j) {           // stage seq chunk 64x256 -> bf16
      int i4 = tid + j * 256;
      int r = i4 >> 6, c4 = i4 & 63;
      v4f v = *(const v4f*)(seqb + (size_t)(sbase + r) * 256 + c4 * 4);
      __bf16* dst = s_stage + r * 264 + c4 * 4;
      dst[0] = (__bf16)v.x; dst[1] = (__bf16)v.y;
      dst[2] = (__bf16)v.z; dst[3] = (__bf16)v.w;
    }
    if (sc < 7)                              // global_prefetch_b8 next chunk
      __builtin_prefetch(seqb + (size_t)(sbase + 64) * 256 + tid * 64, 0, 1);
    __syncthreads();

    v8f acc[2];
    acc[0] = (v8f){0.f,0.f,0.f,0.f,0.f,0.f,0.f,0.f};
    acc[1] = (v8f){0.f,0.f,0.f,0.f,0.f,0.f,0.f,0.f};
    #pragma unroll
    for (int ks = 0; ks < 8; ++ks) {
      v16bf a  = ldsA(s_stage, lane, 264, sstrip * 16, ks * 32);
      v16bf b0 = ldsB(s_tgtw,  lane, 264, (npair + 0) * 16, ks * 32);
      v16bf b1 = ldsB(s_tgtw,  lane, 264, (npair + 1) * 16, ks * 32);
      acc[0] = WMMA_BF16(a, b0, acc[0]);
      acc[1] = WMMA_BF16(a, b1, acc[1]);
    }
    int col = lane & 15, rofs = (lane >> 4) * 8;
    #pragma unroll
    for (int t = 0; t < 2; ++t)
      #pragma unroll
      for (int r = 0; r < 8; ++r)
        s_score[(sbase + sstrip * 16 + rofs + r) * 68 + (npair + t) * 16 + col]
            = acc[t][r];
    __syncthreads();
  }

  // ================= Phase 3: mask + softmax over s, write weight^T ==========
  for (int i = tid; i < 512 * 64; i += 256) {
    int s = i >> 6, n = i & 63;
    if (maskb[s]) s_score[s * 68 + n] = -1e16f;
  }
  __syncthreads();
  const int n3 = tid & 63, q3 = tid >> 6;    // 4 squads x 64 columns
  float m = -3.0e38f;
  for (int s = q3 * 128; s < q3 * 128 + 128; ++s)
    m = fmaxf(m, s_score[s * 68 + n3]);
  s_red[q3 * 64 + n3] = m;
  __syncthreads();
  if (tid < 64)
    s_cmax[tid] = fmaxf(fmaxf(s_red[tid], s_red[64 + tid]),
                        fmaxf(s_red[128 + tid], s_red[192 + tid]));
  __syncthreads();
  const float cm = s_cmax[n3];
  float sum = 0.f;
  for (int s = q3 * 128; s < q3 * 128 + 128; ++s)
    sum += __expf(s_score[s * 68 + n3] - cm);
  s_red[q3 * 64 + n3] = sum;
  __syncthreads();
  if (tid < 64)
    s_csum[tid] = s_red[tid] + s_red[64 + tid] + s_red[128 + tid] + s_red[192 + tid];
  __syncthreads();
  const float inv = 1.0f / s_csum[n3];
  for (int s = q3 * 128; s < q3 * 128 + 128; ++s) {
    float wt = __expf(s_score[s * 68 + n3] - cm) * inv;
    s_wT[n3 * 528 + s] = (__bf16)wt;         // transposed [n][s] for A-fragments
  }
  __syncthreads();

  // ================= Phase 4: all_vec[n][d] = sum_s w[n][s] * seq[s][d] ======
  const int nstrip4 = w & 3;
  const int dblk4   = (w >> 2) * 8;
  v8f acc4[8];
  #pragma unroll
  for (int t = 0; t < 8; ++t) acc4[t] = (v8f){0.f,0.f,0.f,0.f,0.f,0.f,0.f,0.f};

  for (int sc = 0; sc < 8; ++sc) {
    int sbase = sc * 64;
    {  // stage seq chunk TRANSPOSED: LDS seqT [256 d][72] bf16, s in 0..63
      const float* gp = seqb + (size_t)sbase * 256 + tid;   // d = tid
      __bf16* dp = s_stage + tid * 72;
      #pragma unroll 8
      for (int s = 0; s < 64; s += 2) {
        float f0 = gp[(size_t)s * 256];
        float f1 = gp[(size_t)(s + 1) * 256];
        dp[s] = (__bf16)f0; dp[s + 1] = (__bf16)f1;
      }
    }
    if (sc < 7)
      __builtin_prefetch(seqb + (size_t)(sbase + 64) * 256 + tid * 64, 0, 1);
    __syncthreads();
    #pragma unroll
    for (int ks = 0; ks < 2; ++ks) {
      v16bf a = ldsA(s_wT, lane, 528, nstrip4 * 16, sbase + ks * 32);
      #pragma unroll
      for (int t = 0; t < 8; ++t) {
        v16bf bb = ldsB(s_stage, lane, 72, (dblk4 + t) * 16, ks * 32);
        acc4[t] = WMMA_BF16(a, bb, acc4[t]);
      }
    }
    __syncthreads();
  }
  {
    int col = lane & 15, rofs = (lane >> 4) * 8;
    float* ob = out + ((size_t)b * 256 + nb) * 256;
    #pragma unroll
    for (int t = 0; t < 8; ++t)
      #pragma unroll
      for (int r = 0; r < 8; ++r)
        ob[(size_t)(nstrip4 * 16 + rofs + r) * 256 + (dblk4 + t) * 16 + col]
            = acc4[t][r];
  }
}

extern "C" void kernel_launch(void* const* d_in, const int* in_sizes, int n_in,
                              void* d_out, int out_size, void* d_ws, size_t ws_size,
                              hipStream_t stream) {
  (void)in_sizes; (void)n_in; (void)d_ws; (void)ws_size; (void)out_size;
  const float* seq = (const float*)d_in[0];           // [256,512,256] f32
  const float* tgt = (const float*)d_in[1];           // [256,256,256] f32
  const unsigned char* mask = (const unsigned char*)d_in[2];  // [256,512] bool
  const float* W   = (const float*)d_in[3];           // [1,256,256] f32
  float* out = (float*)d_out;                          // [256,256,256] f32

  dim3 grid(4, 256);   // (N/64 tiles, B)
  dim3 block(256);     // 8 wave32
  ta_fused_kernel<<<grid, block, SMEM_BYTES, stream>>>(seq, tgt, mask, W, out);
}
